// HuffmanWeight_90915867722214
// MI455X (gfx1250) — matrix operand order repaired and checked by reading, not compiled, not run on hardware
//
#include <hip/hip_runtime.h>
#include <hip/hip_bf16.h>

#define MDIM   8192
#define NDIM   8192
#define BATCH  8
#define KSPLIT 8                    // waves per block, each owns a K slice
#define KSLICE (NDIM / KSPLIT)      // 1024 -> 32 wmma iterations per wave

typedef __attribute__((ext_vector_type(16))) __bf16       v16bf;
typedef __attribute__((ext_vector_type(8)))  float        v8f;
typedef __attribute__((ext_vector_type(8)))  unsigned int v8u;
typedef __attribute__((ext_vector_type(4)))  int          v4i;
typedef __attribute__((ext_vector_type(4)))  unsigned int v4u;

// Reassemble two bf16 values ((exp<<8)|rem each) into one packed dword.
static __device__ __forceinline__ unsigned int pack2(int r0, int e0, int r1, int e1) {
    unsigned int lo = (unsigned int)r0 | ((unsigned int)e0 << 8);
    unsigned int hi = (unsigned int)r1 | ((unsigned int)e1 << 8);
    return lo | (hi << 16);
}

static __device__ __forceinline__ unsigned short f2bf_bits(float f) {
    __hip_bfloat16 h = __float2bfloat16(f);      // RNE, matches astype(bf16)
    unsigned short u;
    __builtin_memcpy(&u, &h, sizeof(u));
    return u;
}
static __device__ __forceinline__ float bfbits2f(unsigned short u) {
    __hip_bfloat16 h;
    __builtin_memcpy(&h, &u, sizeof(u));
    return __bfloat162float(h);
}

__global__ __launch_bounds__(32 * KSPLIT)
void huffman_gemv_wmma(const __hip_bfloat16* __restrict__ x,
                       const int*            __restrict__ rem,
                       const int*            __restrict__ expb,
                       const float*          __restrict__ bias,
                       __hip_bfloat16*       __restrict__ out) {
    __shared__ float lds[KSPLIT * 256];          // 8 KB: one 16x16 f32 tile per wave

    const int tid  = threadIdx.x;
    const int wave = tid >> 5;
    const int lane = tid & 31;
    const int b    = lane & 15;                  // batch column (valid if < BATCH); also A row-in-tile
    const int hi   = lane >> 4;                  // K-half selector for A/B operand layout
    const int m0   = blockIdx.x * 16;            // this block's 16-row M tile

    const size_t rowoff = (size_t)(m0 + b) * NDIM;   // A row for this lane
    const int    kbase  = wave * KSLICE;

    // Pointer to this lane's x row (batch b); dummy row 0 for padded lanes (loads masked below).
    const unsigned short* xrow = (const unsigned short*)x + (size_t)(b < BATCH ? b : 0) * NDIM;

    v8f acc = {};

    #pragma unroll 2
    for (int k = kbase; k < kbase + KSLICE; k += 32) {
        // ---- A operand: decode 16 bf16 weights (two contiguous runs of 8) ----
        const int kkA = k + hi * 8;
        const v4i* rp = (const v4i*)(rem  + rowoff + kkA);
        const v4i* ep = (const v4i*)(expb + rowoff + kkA);
        // stream-once data: non-temporal so 512MB doesn't thrash L2
        v4i r0 = __builtin_nontemporal_load(rp);         // K = kkA+0..3
        v4i r1 = __builtin_nontemporal_load(rp + 1);     // K = kkA+4..7
        v4i r2 = __builtin_nontemporal_load(rp + 4);     // K = kkA+16..19
        v4i r3 = __builtin_nontemporal_load(rp + 5);     // K = kkA+20..23
        v4i e0 = __builtin_nontemporal_load(ep);
        v4i e1 = __builtin_nontemporal_load(ep + 1);
        v4i e2 = __builtin_nontemporal_load(ep + 4);
        v4i e3 = __builtin_nontemporal_load(ep + 5);

        v8u au;
        au[0] = pack2(r0[0], e0[0], r0[1], e0[1]);
        au[1] = pack2(r0[2], e0[2], r0[3], e0[3]);
        au[2] = pack2(r1[0], e1[0], r1[1], e1[1]);
        au[3] = pack2(r1[2], e1[2], r1[3], e1[3]);
        au[4] = pack2(r2[0], e2[0], r2[1], e2[1]);
        au[5] = pack2(r2[2], e2[2], r2[3], e2[3]);
        au[6] = pack2(r3[0], e3[0], r3[1], e3[1]);
        au[7] = pack2(r3[2], e3[2], r3[3], e3[3]);
        v16bf amat = __builtin_bit_cast(v16bf, au);

        // ---- B operand: x^T slice, two runs of 8 bf16 from row b, same k order as A; zero columns for b>=8 ----
        v4u xlo = {0u, 0u, 0u, 0u};
        v4u xhi = {0u, 0u, 0u, 0u};
        if (b < BATCH) {
            const v4u* xp = (const v4u*)(xrow + k + hi * 8);
            xlo = xp[0];
            xhi = xp[2];
        }
        v8u bu;
        bu[0] = xlo[0]; bu[1] = xlo[1]; bu[2] = xlo[2]; bu[3] = xlo[3];
        bu[4] = xhi[0]; bu[5] = xhi[1]; bu[6] = xhi[2]; bu[7] = xhi[3];
        v16bf bmat = __builtin_bit_cast(v16bf, bu);

        acc = __builtin_amdgcn_wmma_f32_16x16x32_bf16(
                  /*neg_a=*/false, amat, /*neg_b=*/false, bmat,
                  /*c_mod=*/(short)0, acc, /*reuse_a=*/false, /*reuse_b=*/false);
    }

    // ---- cross-wave K reduction through LDS ----
    #pragma unroll
    for (int v = 0; v < 8; ++v)
        lds[wave * 256 + v * 32 + lane] = acc[v];
    __syncthreads();

    {
        float s = 0.f;
        #pragma unroll
        for (int w = 0; w < KSPLIT; ++w) s += lds[w * 256 + tid];
        lds[tid] = s;   // only thread `tid` ever touches lds[tid] in this phase
    }
    __syncthreads();

    // ---- wave 0 finalizes: bf16(acc) + bf16(bias), packed 16B store per lane ----
    if (tid < 32 && b < BATCH) {
        const int mbase = m0 + hi * 8;           // 8 consecutive output rows
        v4u o;
        #pragma unroll
        for (int v2 = 0; v2 < 4; ++v2) {
            const int v = v2 * 2;
            float y0 = bfbits2f(f2bf_bits(lds[(v + 0) * 32 + lane]))
                     + bfbits2f(f2bf_bits(bias[mbase + v + 0]));
            float y1 = bfbits2f(f2bf_bits(lds[(v + 1) * 32 + lane]))
                     + bfbits2f(f2bf_bits(bias[mbase + v + 1]));
            o[v2] = (unsigned int)f2bf_bits(y0) | ((unsigned int)f2bf_bits(y1) << 16);
        }
        *(v4u*)((unsigned short*)out + (size_t)b * MDIM + mbase) = o;
    }
}

extern "C" void kernel_launch(void* const* d_in, const int* in_sizes, int n_in,
                              void* d_out, int out_size, void* d_ws, size_t ws_size,
                              hipStream_t stream) {
    const __hip_bfloat16* x    = (const __hip_bfloat16*)d_in[0];
    const int*            rem  = (const int*)d_in[1];
    const int*            expb = (const int*)d_in[2];
    const float*          bias = (const float*)d_in[3];
    __hip_bfloat16*       out  = (__hip_bfloat16*)d_out;

    dim3 grid(MDIM / 16);         // 512 M-tiles
    dim3 block(32 * KSPLIT);      // 8 waves, each a 1024-wide K slice
    hipLaunchKernelGGL(huffman_gemv_wmma, grid, block, 0, stream,
                       x, rem, expb, bias, out);
}
